// Get_Inters_Property_47588237639972
// MI455X (gfx1250) — matrix operand order repaired and checked by reading, not compiled, not run on hardware
//
#include <hip/hip_runtime.h>

// MI455X / gfx1250, wave32. Fused NeRF MLP: f16 WMMA (v_wmma_f32_16x16x32_f16),
// f32 accumulate. Weights pre-packed (fp32 -> f16, fragment-transposed) once per
// launch into d_ws; main kernel streams 16KB weight slabs into a double-buffered
// LDS tile with GLOBAL_LOAD_ASYNC_TO_LDS_B128 (ASYNCcnt) when available.

typedef _Float16 v8h  __attribute__((ext_vector_type(8)));
typedef _Float16 v16h __attribute__((ext_vector_type(16)));
typedef float    v8f  __attribute__((ext_vector_type(8)));
typedef int      i32x4 __attribute__((__vector_size__(16)));   // matches builtin param type

#define TPB    256
#define BM     32
#define NROWS  (8192 * 32)
#define SLABH  8192              // halves per packed slab: 256 cols x 32 k

#ifndef __has_builtin
#define __has_builtin(x) 0
#endif
#if __has_builtin(__builtin_amdgcn_global_load_async_to_lds_b128)
#define HAVE_ASYNC_LDS 1
#else
#define HAVE_ASYNC_LDS 0
#endif

#define AS1 __attribute__((address_space(1)))
#define AS3 __attribute__((address_space(3)))

// ---------------- WMMA fragment loaders (LDS -> VGPR) ----------------
// A-matrix 16x32 f16 layout (ISA 7.12.2): lane l = (hs = l>>4, m = l&15):
// halves 0..7 = K[hs*8 ..), halves 8..15 = K[16+hs*8 ..)
__device__ __forceinline__ v16h load_frag_a(const _Float16* p) {
  v8h lo = *(const v8h*)(p);
  v8h hi = *(const v8h*)(p + 16);
  return __builtin_shufflevector(lo, hi, 0,1,2,3,4,5,6,7,8,9,10,11,12,13,14,15);
}
// B-matrix 32x16 f16: lane l = (hs, n = l&15): halves 0..7 = K[hs*8 ..), halves 8..15 = K[16+hs*8 ..)
__device__ __forceinline__ v16h load_frag_b(const _Float16* p) {
  v8h lo = *(const v8h*)(p);
  v8h hi = *(const v8h*)(p + 16);
  return __builtin_shufflevector(lo, hi, 0,1,2,3,4,5,6,7,8,9,10,11,12,13,14,15);
}

// ---------------- weight slab copy: packed global f16 -> LDS ----------------
__device__ __forceinline__ void slab_copy_sync(const _Float16* __restrict__ src,
                                               _Float16* __restrict__ dst, int tid) {
  const uint4* s4 = (const uint4*)src;
  uint4*       d4 = (uint4*)dst;
  #pragma unroll
  for (int i = 0; i < 4; ++i) d4[tid + i * TPB] = s4[tid + i * TPB];
}

__device__ __forceinline__ void slab_copy(const _Float16* src, _Float16* dst, int tid) {
#if HAVE_ASYNC_LDS
  #pragma unroll
  for (int i = 0; i < 4; ++i) {
    __builtin_amdgcn_global_load_async_to_lds_b128(
        (AS1 i32x4*)(src + (size_t)(tid + i * TPB) * 8),
        (AS3 i32x4*)(dst + (size_t)(tid + i * TPB) * 8),
        /*offset=*/0, /*cpol=*/0);
  }
#else
  slab_copy_sync(src, dst, tid);
#endif
}

__device__ __forceinline__ void async_wait0() {
#if HAVE_ASYNC_LDS
#if __has_builtin(__builtin_amdgcn_s_wait_asynccnt)
  __builtin_amdgcn_s_wait_asynccnt(0);
#else
  asm volatile("s_wait_asynccnt 0x0" ::: "memory");
#endif
#endif
}

// ---------------- one dense layer ----------------
// Y[BM,256] = act(X[BM, nslab*32] @ Wpacked + b); Wpacked = [nslab][256][32] f16.
// Double-buffered LDS weight slab; async copy of slab s+1 overlaps WMMA on slab s.
__device__ void layer_gemm(const _Float16* __restrict__ Wh, const float* __restrict__ bg,
                           int nslab,
                           const _Float16* X, int xs,
                           _Float16* Y, int ys,
                           _Float16* wbuf /*[2][SLABH]*/, int relu_en, int tid) {
  const int lane = tid & 31;
  const int wave = tid >> 5;
  const int mt   = wave >> 2;        // 0..1   : M tile (16 rows)
  const int ntb  = (wave & 3) * 4;   // 0,4,8,12: first of 4 N tiles (16 cols)
  const int hs   = lane >> 4;
  const int l15  = lane & 15;

  v8f acc[4] = {};

  slab_copy(Wh, wbuf, tid);          // prologue: slab 0 -> buf 0
  async_wait0();
  __syncthreads();

  for (int s = 0; s < nslab; ++s) {
    const _Float16* cur = wbuf + (s & 1) * SLABH;
    if (s + 1 < nslab)               // prefetch next slab into other buffer
      slab_copy(Wh + (size_t)(s + 1) * SLABH, wbuf + ((s + 1) & 1) * SLABH, tid);

    v16h afrag = load_frag_a(X + (mt * 16 + l15) * xs + s * 32 + hs * 8);
    #pragma unroll
    for (int t = 0; t < 4; ++t) {
      v16h b = load_frag_b(cur + ((ntb + t) * 16 + l15) * 32 + hs * 8);
      acc[t] = __builtin_amdgcn_wmma_f32_16x16x32_f16(
          false, afrag, false, b, (short)0, acc[t], false, false);
    }
    async_wait0();
    __syncthreads();
  }

  // epilogue: bias + activation, f16 store to LDS (C/D layout: col=l15, m=hs*8+r)
  #pragma unroll
  for (int t = 0; t < 4; ++t) {
    int col  = (ntb + t) * 16 + l15;
    float bv = bg[col];
    #pragma unroll
    for (int r = 0; r < 8; ++r) {
      int m   = mt * 16 + hs * 8 + r;
      float v = acc[t][r] + bv;
      if (relu_en) v = fmaxf(v, 0.0f);
      Y[m * ys + col] = (_Float16)v;
    }
  }
}

// ---------------- unguarded staging helpers ----------------
__device__ __forceinline__ void stage_rows(const float* __restrict__ G, int C,
                                           long grow0, _Float16* dst, int ds,
                                           int dstoff, int tid) {
  for (int idx = tid; idx < BM * C; idx += TPB) {
    int row = idx / C;
    int c   = idx - row * C;
    dst[row * ds + dstoff + c] = (_Float16)G[(grow0 + row) * (long)C + c];
  }
}

__device__ __forceinline__ void copy_lds(const _Float16* src, int ss,
                                         _Float16* dst, int ds, int dstoff,
                                         int C, int tid) {
  for (int idx = tid; idx < BM * C; idx += TPB) {
    int row = idx / C;
    int c   = idx - row * C;
    dst[row * ds + dstoff + c] = src[row * ss + c];
  }
}

// ---------------- weight pre-pack: fp32 [K,256] -> f16 [nslab][256][32] ----------------
__global__ void pack_weights_kernel(const float* __restrict__ Wg, int Kreal,
                                    _Float16* __restrict__ dst) {
  const int s = blockIdx.x;        // slab
  const int n = threadIdx.x;       // output feature 0..255
  _Float16 tmp[32];
  #pragma unroll
  for (int k = 0; k < 32; ++k) {
    int kg = s * 32 + k;
    tmp[k] = (kg < Kreal) ? (_Float16)Wg[(size_t)kg * 256 + n] : (_Float16)0.0f;
  }
  v8h* out = (v8h*)(dst + (size_t)s * SLABH + n * 32);
  const v8h* t = (const v8h*)tmp;
  out[0] = t[0]; out[1] = t[1]; out[2] = t[2]; out[3] = t[3];
}

// ---------------- fused kernel ----------------
__global__ void __launch_bounds__(TPB)
nerf_fused_kernel(const float* __restrict__ pos, const float* __restrict__ dir,
                  const float* __restrict__ app, const float* __restrict__ gemo,
                  const float* __restrict__ color, const float* __restrict__ mask,
                  const _Float16* __restrict__ P0, const float* __restrict__ b0,
                  const _Float16* __restrict__ P1, const float* __restrict__ b1,
                  const _Float16* __restrict__ P2, const float* __restrict__ b2,
                  const float* __restrict__ Ws, const float* __restrict__ bs,
                  const _Float16* __restrict__ P3, const float* __restrict__ b3,
                  const _Float16* __restrict__ P4, const float* __restrict__ b4,
                  const float* __restrict__ Wr, const float* __restrict__ br,
                  float* __restrict__ out_sigma, float* __restrict__ out_rgb) {
  extern __shared__ _Float16 smem[];
  _Float16* bufA = smem;                  // [32][512]  layer input (concat space)
  _Float16* bufB = bufA + BM * 512;       // [32][256]  d1, later d4
  _Float16* bufC = bufB + BM * 256;       // [32][256]  d0/d2/d3
  _Float16* wbuf = bufC + BM * 256;       // [2][SLABH] double-buffered weight slab

  const int  tid   = threadIdx.x;
  const long grow0 = (long)blockIdx.x * BM;

  // ---- L0: d0 = relu(pos @ W0 + b0), K = 63 (pad 64)
  stage_rows(pos, 63, grow0, bufA, 512, 0, tid);
  if (tid < BM) bufA[tid * 512 + 63] = (_Float16)0.0f;   // pad col
  __syncthreads();
  layer_gemm(P0, b0, 2, bufA, 512, bufC, 256, wbuf, 1, tid);
  __syncthreads();

  // ---- L1: d1 = relu(d0 @ W1 + b1), K = 256
  layer_gemm(P1, b1, 8, bufC, 256, bufB, 256, wbuf, 1, tid);
  __syncthreads();

  // ---- L2: d2 = relu([d1 | gemo] @ W2 + b2), K = 512
  copy_lds(bufB, 256, bufA, 512, 0, 256, tid);           // d1 -> bufA[:,0:256]
  stage_rows(gemo, 256, grow0, bufA, 512, 256, tid);
  __syncthreads();
  layer_gemm(P2, b2, 16, bufA, 512, bufC, 256, wbuf, 1, tid);
  __syncthreads();

  // ---- sigma head: softplus(d2 @ Ws + bs) * mask  (d2 in bufC)
  if (tid < BM) {
    long  gr = grow0 + tid;
    float a  = bs[0];
    for (int k = 0; k < 256; ++k) a += (float)bufC[tid * 256 + k] * Ws[k];
    float sp = (a > 20.0f) ? a : log1pf(__expf(a));
    out_sigma[gr] = sp * mask[gr];
  }

  // ---- L3: d3 = relu([d1 | color] @ W3 + b3); d1 still in bufA[:,0:256]
  stage_rows(color, 256, grow0, bufA, 512, 256, tid);
  __syncthreads();
  layer_gemm(P3, b3, 16, bufA, 512, bufC, 256, wbuf, 1, tid);
  __syncthreads();

  // ---- L4: d4 = relu([d3 | dir | app | 0pad] @ W4 + b4), K = 331 (pad 352)
  copy_lds(bufC, 256, bufA, 512, 0, 256, tid);           // d3 -> bufA[:,0:256]
  stage_rows(dir, 27, grow0, bufA, 512, 256, tid);       // cols 256..282
  for (int idx = tid; idx < BM * 48; idx += TPB) {       // cols 283..330 (broadcast app)
    int row = idx / 48;
    int c   = idx - row * 48;
    bufA[row * 512 + 283 + c] = (_Float16)app[c];
  }
  for (int idx = tid; idx < BM * 21; idx += TPB) {       // cols 331..351 zero pad
    int row = idx / 21;
    int c   = idx - row * 21;
    bufA[row * 512 + 331 + c] = (_Float16)0.0f;
  }
  __syncthreads();
  layer_gemm(P4, b4, 11, bufA, 512, bufB, 256, wbuf, 1, tid);
  __syncthreads();

  // ---- rgb head: sigmoid(d4 @ Wr + br)  (d4 in bufB)
  if (tid < BM) {
    long  gr = grow0 + tid;
    float r0 = br[0], r1 = br[1], r2 = br[2];
    for (int k = 0; k < 256; ++k) {
      float a = (float)bufB[tid * 256 + k];
      r0 += a * Wr[k * 3 + 0];
      r1 += a * Wr[k * 3 + 1];
      r2 += a * Wr[k * 3 + 2];
    }
    out_rgb[gr * 3 + 0] = 1.0f / (1.0f + __expf(-r0));
    out_rgb[gr * 3 + 1] = 1.0f / (1.0f + __expf(-r1));
    out_rgb[gr * 3 + 2] = 1.0f / (1.0f + __expf(-r2));
  }
}

extern "C" void kernel_launch(void* const* d_in, const int* in_sizes, int n_in,
                              void* d_out, int out_size, void* d_ws, size_t ws_size,
                              hipStream_t stream) {
  const float* pos   = (const float*)d_in[0];
  const float* dir   = (const float*)d_in[1];
  const float* app   = (const float*)d_in[2];
  const float* gemo  = (const float*)d_in[3];
  const float* color = (const float*)d_in[4];
  const float* mask  = (const float*)d_in[5];
  // d_in[6] = num_inters (int, fixed 32)
  const float* W0 = (const float*)d_in[7];
  const float* b0 = (const float*)d_in[8];
  const float* W1 = (const float*)d_in[9];
  const float* b1 = (const float*)d_in[10];
  const float* W2 = (const float*)d_in[11];
  const float* b2 = (const float*)d_in[12];
  const float* Ws = (const float*)d_in[13];
  const float* bs = (const float*)d_in[14];
  const float* W3 = (const float*)d_in[15];
  const float* b3 = (const float*)d_in[16];
  const float* W4 = (const float*)d_in[17];
  const float* b4 = (const float*)d_in[18];
  const float* Wr = (const float*)d_in[19];
  const float* br = (const float*)d_in[20];

  // packed f16 weights in workspace: [slab][256][32]; slabs: 2,8,16,16,11 (53 total, ~848KB)
  _Float16* P0 = (_Float16*)d_ws;
  _Float16* P1 = P0 + (size_t)2  * SLABH;
  _Float16* P2 = P1 + (size_t)8  * SLABH;
  _Float16* P3 = P2 + (size_t)16 * SLABH;
  _Float16* P4 = P3 + (size_t)16 * SLABH;

  pack_weights_kernel<<<dim3(2),  dim3(TPB), 0, stream>>>(W0, 63,  P0);
  pack_weights_kernel<<<dim3(8),  dim3(TPB), 0, stream>>>(W1, 256, P1);
  pack_weights_kernel<<<dim3(16), dim3(TPB), 0, stream>>>(W2, 512, P2);
  pack_weights_kernel<<<dim3(16), dim3(TPB), 0, stream>>>(W3, 512, P3);
  pack_weights_kernel<<<dim3(11), dim3(TPB), 0, stream>>>(W4, 331, P4);

  float* out_sigma = (float*)d_out;        // [8192*32]
  float* out_rgb   = out_sigma + NROWS;    // [8192*32*3]

  const size_t lds_bytes =
      (size_t)(BM * 512 + BM * 256 + BM * 256 + 2 * SLABH) * sizeof(_Float16); // 96 KB

  nerf_fused_kernel<<<dim3(NROWS / BM), dim3(TPB), lds_bytes, stream>>>(
      pos, dir, app, gemo, color, mask,
      P0, b0, P1, b1, P2, b2, Ws, bs, P3, b3, P4, b4, Wr, br,
      out_sigma, out_rgb);
}
